// ScaledDotProductAttention_30734785970418
// MI455X (gfx1250) — matrix operand order repaired and checked by reading, not compiled, not run on hardware
//
#include <hip/hip_runtime.h>
#include <hip/hip_bf16.h>

typedef __attribute__((ext_vector_type(16))) _Float16 v16h;
typedef __attribute__((ext_vector_type(8)))  _Float16 v8h;
typedef __attribute__((ext_vector_type(8)))  float    v8f;

// ---------------------------------------------------------------------------
// WMMA helpers (CDNA5: V_WMMA_F32_16X16X32_F16, wave32)
// ---------------------------------------------------------------------------
__device__ __forceinline__ v8f wmma32(v16h a, v16h b, v8f c) {
  // (neg_a, A, neg_b, B, c_mod, C, reuse_a, reuse_b)
  return __builtin_amdgcn_wmma_f32_16x16x32_f16(false, a, false, b, (short)0, c,
                                                false, false);
}

__device__ __forceinline__ v8f zero8() {
  v8f z = {0.f, 0.f, 0.f, 0.f, 0.f, 0.f, 0.f, 0.f};
  return z;
}

// A-matrix 16x32 fragment (MxK, 16-bit). Lane l holds row M = l&15.
// lanes 0-15: K = 0..15 contiguous; lanes 16-31: K = 16..31 contiguous.
// klo = (lane>>4)*8.  rowptr must be 16B aligned.
__device__ __forceinline__ v16h load_a_frag(const _Float16* rowptr, int klo) {
  v8h lo = *(const v8h*)(rowptr + 2 * klo);
  v8h hi = *(const v8h*)(rowptr + 2 * klo + 8);
  v16h r;
#pragma unroll
  for (int i = 0; i < 8; ++i) { r[i] = lo[i]; r[i + 8] = hi[i]; }
  return r;
}

// B-matrix 32x16 fragment (KxN, 16-bit). Lane l holds column N = l&15,
// K = (lane>>4)*16 + 0..15 contiguous.  khi16 = (lane>>4)*16.
// colptr must be 32B aligned.
__device__ __forceinline__ v16h load_b_frag(const _Float16* colptr, int khi16) {
  return *(const v16h*)(colptr + khi16);
}

// ---------------------------------------------------------------------------
// fp32 -> f16 conversion (grid-stride)
// ---------------------------------------------------------------------------
__global__ void f2h_kernel(const float* __restrict__ src,
                           _Float16* __restrict__ dst, int n) {
  for (int i = blockIdx.x * blockDim.x + threadIdx.x; i < n;
       i += gridDim.x * blockDim.x)
    dst[i] = (_Float16)src[i];
}

// ---------------------------------------------------------------------------
// C[m, e] = sum_k A[m, k] * W[e, k] + bias[e]     (M=4096, N=K=1024)
// A row-major MxK f16, W row-major NxK f16 (nn.Linear layout).
// One wave -> 16x64 tile (4 accumulators, shared A fragment).
// Pipeline unrolled x2 with ping-pong fragment buffers: no rotation copies,
// loads for step k+32 overlap the WMMAs of step k.
// Block: 256 threads = 8 waves = 128 rows.  Grid: (32, 16).
// mode 0: store f16 at [((b*16+h)*2048+n)*64+d]      (Q / K layout)
// mode 1: store f16 at [((b*16+h)*64+d)*2048+n]      (V transposed)
// mode 2: store f32 at [m*1024+e]                    (final output)
// ---------------------------------------------------------------------------
__global__ __launch_bounds__(256) void gemm1024_wmma(
    const _Float16* __restrict__ A, const _Float16* __restrict__ W,
    const float* __restrict__ bias, void* __restrict__ out, int mode) {
  const int lane  = threadIdx.x & 31;
  const int wave  = threadIdx.x >> 5;
  const int m0    = blockIdx.x * 128 + wave * 16;
  const int n0    = blockIdx.y * 64;
  const int r15   = lane & 15;
  const int klo   = (lane >> 4) * 8;
  const int khi16 = (lane >> 4) * 16;

  const _Float16* arow  = A + (size_t)(m0 + r15) * 1024;
  const _Float16* wrow0 = W + (size_t)(n0 + r15) * 1024;

  v8f acc[4];
#pragma unroll
  for (int j = 0; j < 4; ++j) acc[j] = zero8();

  v16h aA, aB;
  v16h bA[4], bB[4];

  auto load_set = [&](int kt, v16h& a, v16h (&b)[4]) {
    a = load_a_frag(arow + kt, klo);
#pragma unroll
    for (int j = 0; j < 4; ++j)
      b[j] = load_b_frag(wrow0 + (size_t)j * 16 * 1024 + kt, khi16);
  };
  auto compute = [&](const v16h& a, const v16h (&b)[4]) {
#pragma unroll
    for (int j = 0; j < 4; ++j) acc[j] = wmma32(a, b[j], acc[j]);
  };

  // ---- prologue: step 0 into buffer A ----
  load_set(0, aA, bA);

  // ---- steady state: 2 steps per iteration, ping-pong A/B ----
  for (int kt = 0; kt < 1024 - 64; kt += 64) {
    load_set(kt + 32, aB, bB);   // next step into B
    compute(aA, bA);             // consume A (loads for B in flight)
    load_set(kt + 64, aA, bA);   // step after next into A
    compute(aB, bB);             // consume B
  }
  // ---- epilogue: steps 30 (in A) and 31 ----
  load_set(1024 - 32, aB, bB);
  compute(aA, bA);
  compute(aB, bB);

#pragma unroll
  for (int j = 0; j < 4; ++j) {
    const int e  = n0 + j * 16 + r15;      // output column for this lane
    const float bv = bias[e];
    const int hh = e >> 6, dd = e & 63;
#pragma unroll
    for (int r = 0; r < 8; ++r) {
      const int m  = m0 + r + ((lane >> 4) << 3);  // output row (C layout)
      const float v = acc[j][r] + bv;
      if (mode == 2) {
        ((float*)out)[(size_t)m * 1024 + e] = v;
      } else {
        const int bb = m >> 11, nn = m & 2047;
        const size_t idx =
            (mode == 0) ? ((size_t)((bb * 16 + hh) * 2048 + nn) * 64 + dd)
                        : ((size_t)((bb * 16 + hh) * 64 + dd) * 2048 + nn);
        ((_Float16*)out)[idx] = (_Float16)v;
      }
    }
  }
}

// ---------------------------------------------------------------------------
// Flash attention: one block = one (b,h) pair x 128 query rows (8 waves x 16).
// Per 32-key step: all K/V fragment loads are issued up front; the softmax
// VALU work (exp/max/shuffles) covers the V-load latency before the P*V
// WMMAs.  P is staged through per-wave LDS to reach A-fragment layout.
// Q/K: [b*16+h][n][64] f16;  VT: [b*16+h][64][n] f16;  O: [b][n][h*64+d] f16.
// ---------------------------------------------------------------------------
__global__ __launch_bounds__(256) void attn_wmma(
    const _Float16* __restrict__ Q, const _Float16* __restrict__ K,
    const _Float16* __restrict__ VT, _Float16* __restrict__ O) {
  __shared__ __align__(128) _Float16 plds[8][16][32];

  const int lane  = threadIdx.x & 31;
  const int wave  = threadIdx.x >> 5;
  const int bh    = blockIdx.y;                 // b*16 + h
  const int q0    = blockIdx.x * 128 + wave * 16;
  const int r15   = lane & 15;
  const int hi8   = (lane >> 4) * 8;
  const int klo   = hi8;
  const int khi16 = hi8 * 2;

  const _Float16* qrow = Q + ((size_t)bh * 2048 + q0 + r15) * 64;
  v16h qa0 = load_a_frag(qrow, klo);            // d = 0..31
  v16h qa1 = load_a_frag(qrow + 32, klo);       // d = 32..63

  const _Float16* kbase = K + (size_t)bh * 2048 * 64;
  const _Float16* vbase = VT + (size_t)bh * 64 * 2048;

  v8f o[4];
#pragma unroll
  for (int j = 0; j < 4; ++j) o[j] = zero8();
  float mrow[8], lrow[8];
#pragma unroll
  for (int r = 0; r < 8; ++r) { mrow[r] = -3.0e38f; lrow[r] = 0.f; }

  _Float16(*P)[32] = plds[wave];

  for (int kk = 0; kk < 2048; kk += 32) {
    // ---- issue ALL global loads for this step up front ----
    const _Float16* krow0 = kbase + (size_t)(kk + r15) * 64;   // keys kk..+15
    const _Float16* krow1 = krow0 + 16 * 64;                   // keys kk+16..+31
    v16h kb00 = load_b_frag(krow0, khi16);
    v16h kb01 = load_b_frag(krow0 + 32, khi16);
    v16h kb10 = load_b_frag(krow1, khi16);
    v16h kb11 = load_b_frag(krow1 + 32, khi16);
    v16h vb[4];
#pragma unroll
    for (int j = 0; j < 4; ++j) {
      const _Float16* vrow = vbase + (size_t)(j * 16 + r15) * 2048 + kk;
      vb[j] = load_b_frag(vrow, khi16);
    }
    // prefetch next key-step's K lines into near cache (global_prefetch_b8)
    if (kk + 32 < 2048) {
      __builtin_prefetch(krow0 + 32 * 64, 0, 3);
      __builtin_prefetch(krow1 + 32 * 64, 0, 3);
    }

    // ---- S = (Q K^T) for keys [kk, kk+32) ----
    v8f s0 = wmma32(qa0, kb00, zero8());
    s0     = wmma32(qa1, kb01, s0);
    v8f s1 = wmma32(qa0, kb10, zero8());
    s1     = wmma32(qa1, kb11, s1);

    // ---- online softmax (row M = r + hi8, cols across 16 lanes) ----
#pragma unroll
    for (int r = 0; r < 8; ++r) {
      float a = s0[r] * 0.125f;                 // 1/sqrt(64)
      float b = s1[r] * 0.125f;
      float mx = fmaxf(a, b);
      mx = fmaxf(mx, __shfl_xor(mx, 1, 32));
      mx = fmaxf(mx, __shfl_xor(mx, 2, 32));
      mx = fmaxf(mx, __shfl_xor(mx, 4, 32));
      mx = fmaxf(mx, __shfl_xor(mx, 8, 32));    // row max of this 32-key tile
      const float mnew  = fmaxf(mrow[r], mx);
      const float alpha = __expf(mrow[r] - mnew);
      const float p0 = __expf(a - mnew);
      const float p1 = __expf(b - mnew);
      float rs = p0 + p1;
      rs += __shfl_xor(rs, 1, 32);
      rs += __shfl_xor(rs, 2, 32);
      rs += __shfl_xor(rs, 4, 32);
      rs += __shfl_xor(rs, 8, 32);              // row sum of exp
      lrow[r] = lrow[r] * alpha + rs;
      mrow[r] = mnew;
      o[0][r] *= alpha; o[1][r] *= alpha; o[2][r] *= alpha; o[3][r] *= alpha;
      P[r + hi8][r15]      = (_Float16)p0;      // key col kk + r15
      P[r + hi8][r15 + 16] = (_Float16)p1;      // key col kk + 16 + r15
    }

    // ---- O += P * V  (P as A-fragment via LDS; V^T rows are contiguous) ----
    v16h pa = load_a_frag(&P[r15][0], klo);
#pragma unroll
    for (int j = 0; j < 4; ++j) o[j] = wmma32(pa, vb[j], o[j]);
  }

  // ---- normalize and store: O[b][n][h*64 + d] ----
  const int b = bh >> 4, h = bh & 15;
#pragma unroll
  for (int r = 0; r < 8; ++r) {
    const float inv = 1.0f / lrow[r];
    const int n = q0 + r + hi8;
    _Float16* op = O + ((size_t)(b * 2048 + n)) * 1024 + h * 64 + r15;
#pragma unroll
    for (int j = 0; j < 4; ++j) op[j * 16] = (_Float16)(o[j][r] * inv);
  }
}

// ---------------------------------------------------------------------------
// host-side orchestration
// ---------------------------------------------------------------------------
extern "C" void kernel_launch(void* const* d_in, const int* in_sizes, int n_in,
                              void* d_out, int out_size, void* d_ws,
                              size_t ws_size, hipStream_t stream) {
  const float* y  = (const float*)d_in[0];
  const float* Wq = (const float*)d_in[1];
  const float* bq = (const float*)d_in[2];
  const float* Wk = (const float*)d_in[3];
  const float* bk = (const float*)d_in[4];
  const float* Wv = (const float*)d_in[5];
  const float* bv = (const float*)d_in[6];
  const float* Wo = (const float*)d_in[7];
  const float* bo = (const float*)d_in[8];

  const size_t MTOK = 4096;          // B*N
  const size_t DM   = 1024;          // D_MODEL

  _Float16* ws  = (_Float16*)d_ws;
  _Float16* yh  = ws;                      // 4096*1024
  _Float16* wqh = yh  + MTOK * DM;         // 1024*1024
  _Float16* wkh = wqh + DM * DM;
  _Float16* wvh = wkh + DM * DM;
  _Float16* woh = wvh + DM * DM;
  _Float16* qh  = woh + DM * DM;           // [bh][n][64]
  _Float16* kh  = qh  + MTOK * DM;
  _Float16* vth = kh  + MTOK * DM;         // [bh][64][n]
  _Float16* ohh = vth + MTOK * DM;         // [b][n][1024]
  // total: 48 MiB of workspace

  f2h_kernel<<<1024, 256, 0, stream>>>(y,  yh,  (int)(MTOK * DM));
  f2h_kernel<<<256,  256, 0, stream>>>(Wq, wqh, (int)(DM * DM));
  f2h_kernel<<<256,  256, 0, stream>>>(Wk, wkh, (int)(DM * DM));
  f2h_kernel<<<256,  256, 0, stream>>>(Wv, wvh, (int)(DM * DM));
  f2h_kernel<<<256,  256, 0, stream>>>(Wo, woh, (int)(DM * DM));

  dim3 gg(32, 16);                   // M/128, N/64
  gemm1024_wmma<<<gg, 256, 0, stream>>>(yh, wqh, bq, qh,  0);
  gemm1024_wmma<<<gg, 256, 0, stream>>>(yh, wkh, bk, kh,  0);
  gemm1024_wmma<<<gg, 256, 0, stream>>>(yh, wvh, bv, vth, 1);

  attn_wmma<<<dim3(16, 32), 256, 0, stream>>>(qh, kh, vth, ohh);

  gemm1024_wmma<<<gg, 256, 0, stream>>>(ohh, woh, bo, d_out, 2);
}
